// Attention_21285857919430
// MI455X (gfx1250) — matrix operand order repaired and checked by reading, not compiled, not run on hardware
//
#include <hip/hip_runtime.h>
#include <hip/hip_bf16.h>

// Problem constants
#define C_DIM 512
#define N_SP  4096           // H*W
#define B_DIM 16
#define G_DIM 32
#define M_TOT (B_DIM * N_SP) // 65536 rows of the flattened (b*n, c) activations
#define EPS_GN 1e-5f
#define SCORE_SCALE 0.044194173824159216f  // 1/sqrt(512)

// GEMM tiling: block tile 256x128, K-step 64, 8 waves of 64x64 each.
#define BM 256
#define BN 128
#define BK 64

typedef __attribute__((ext_vector_type(16))) __bf16 bf16x16;
typedef __attribute__((ext_vector_type(8)))  __bf16 bf16x8;
typedef __attribute__((ext_vector_type(8)))  float  f32x8;

// ---------------------------------------------------------------------------
// WMMA helpers (CDNA5 v_wmma_f32_16x16x32_bf16, wave32)
// ---------------------------------------------------------------------------
__device__ __forceinline__ f32x8 wmma_bf16(bf16x16 a, bf16x16 b, f32x8 c) {
  return __builtin_amdgcn_wmma_f32_16x16x32_bf16(
      /*neg_a=*/false, a, /*neg_b=*/false, b,
      /*c_mod=*/(short)0, c, /*reuse_a=*/false, /*reuse_b=*/false);
}

// A-fragment (16x32, 16-bit): lane m = lane&15, half = lane>>4.
// vector elems 0..7  = K[ks + half*8 .. +7], elems 8..15 = K[ks+16+half*8 .. +7]
// LDS A tile is row-major [BM][BK] bf16.
__device__ __forceinline__ bf16x16 frag_a(const __bf16* tile, int m_off, int ks, int lane) {
  const int m = m_off + (lane & 15);
  const int h = lane >> 4;
  const __bf16* base = tile + m * BK + ks;
  bf16x8 lo = *(const bf16x8*)(base + h * 8);
  bf16x8 hi = *(const bf16x8*)(base + 16 + h * 8);
  return __builtin_shufflevector(lo, hi, 0,1,2,3,4,5,6,7,8,9,10,11,12,13,14,15);
}

// B-fragment (32x16, 16-bit): lane n = lane&15, half = lane>>4,
// vector elems 0..7 = K[ks + half*8 .. +7], elems 8..15 = K[ks+16+half*8 .. +7]. LDS B tile is N-major: [BN][BK] bf16.
__device__ __forceinline__ bf16x16 frag_b(const __bf16* tile, int n_off, int ks, int lane) {
  const int n = n_off + (lane & 15);
  const int h = lane >> 4;
  const __bf16* base = tile + n * BK + ks;
  bf16x8 lo = *(const bf16x8*)(base + h * 8);
  bf16x8 hi = *(const bf16x8*)(base + 16 + h * 8);
  return __builtin_shufflevector(lo, hi, 0,1,2,3,4,5,6,7,8,9,10,11,12,13,14,15);
}

// Core compute on one staged K-round: 8 A-frags + 8 B-frags -> 32 WMMAs/wave.
__device__ __forceinline__ void gemm_round(const __bf16* sA, const __bf16* sB,
                                           int wm, int wn, int lane,
                                           f32x8 acc[4][4]) {
#pragma unroll
  for (int ks = 0; ks < BK; ks += 32) {
    bf16x16 a[4], b[4];
#pragma unroll
    for (int i = 0; i < 4; ++i) a[i] = frag_a(sA, wm + i * 16, ks, lane);
#pragma unroll
    for (int j = 0; j < 4; ++j) b[j] = frag_b(sB, wn + j * 16, ks, lane);
#pragma unroll
    for (int i = 0; i < 4; ++i)
#pragma unroll
      for (int j = 0; j < 4; ++j)
        acc[i][j] = wmma_bf16(a[i], b[j], acc[i][j]);
  }
}

// Stage A: BM x BK tile from row-major src. One thread copies one 128B row.
__device__ __forceinline__ void stage_a(__bf16* sA, const __bf16* gA, int lda, int tid) {
  const __bf16* src = gA + (size_t)tid * lda;
  __bf16* dst = sA + tid * BK;
#pragma unroll
  for (int j = 0; j < 4; ++j)
    *(bf16x16*)(dst + j * 16) = *(const bf16x16*)(src + j * 16);
}

// Stage B (operand already N-major, contiguous in K): BN x BK tile.
__device__ __forceinline__ void stage_b_rows(__bf16* sB, const __bf16* gB, int ldb, int tid) {
  const int r = tid >> 1, c = (tid & 1) * 32;
  const __bf16* src = gB + (size_t)r * ldb + c;
  __bf16* dst = sB + r * BK + c;
  *(bf16x16*)(dst)      = *(const bf16x16*)(src);
  *(bf16x16*)(dst + 16) = *(const bf16x16*)(src + 16);
}

// Stage B with transpose (operand is K-major, contiguous in N): BK x BN tile.
__device__ __forceinline__ void stage_b_transpose(__bf16* sB, const __bf16* gB, int ldb, int tid) {
  const int k = tid >> 2, n = (tid & 3) * 32;
  const __bf16* src = gB + (size_t)k * ldb + n;
#pragma unroll
  for (int j = 0; j < 4; ++j) {
    bf16x8 v = *(const bf16x8*)(src + j * 8);
#pragma unroll
    for (int i = 0; i < 8; ++i) sB[(n + j * 8 + i) * BK + k] = v[i];
  }
}

// ---------------------------------------------------------------------------
// Kernel 0: pack Wq/Wk/Wv/Wo (f32, row-major C x C) into one bf16 buffer
// ---------------------------------------------------------------------------
__global__ void __launch_bounds__(256)
cvt_w_kernel(const float* __restrict__ wq, const float* __restrict__ wk,
             const float* __restrict__ wv, const float* __restrict__ wo,
             __bf16* __restrict__ wbf) {
  const int i = blockIdx.x * 256 + threadIdx.x;       // 0 .. 4*512*512-1
  const int m = i >> 18;                              // which matrix
  const int j = i & ((1 << 18) - 1);
  const float* s = (m == 0) ? wq : (m == 1) ? wk : (m == 2) ? wv : wo;
  wbf[i] = (__bf16)s[j];
}

// ---------------------------------------------------------------------------
// Kernel 1: GroupNorm statistics. One block per (b, g); the group's
// 16 channels * 4096 spatial = 65536 floats are CONTIGUOUS in x.
// ---------------------------------------------------------------------------
__global__ void __launch_bounds__(256)
gn_stats_kernel(const float* __restrict__ x, float* __restrict__ mean,
                float* __restrict__ rstd) {
  const int bg = blockIdx.x;                          // b*32 + g, 0..511
  const float4* p = (const float4*)(x + (size_t)bg * 65536);
  float s = 0.f, ss = 0.f;
  for (int i = threadIdx.x; i < 16384; i += 256) {
    float4 v = p[i];
    s  += v.x + v.y + v.z + v.w;
    ss += v.x * v.x + v.y * v.y + v.z * v.z + v.w * v.w;
  }
  __shared__ float rs[256];
  __shared__ float rq[256];
  rs[threadIdx.x] = s; rq[threadIdx.x] = ss;
  __syncthreads();
  for (int o = 128; o > 0; o >>= 1) {
    if (threadIdx.x < o) { rs[threadIdx.x] += rs[threadIdx.x + o];
                           rq[threadIdx.x] += rq[threadIdx.x + o]; }
    __syncthreads();
  }
  if (threadIdx.x == 0) {
    float m   = rs[0] * (1.f / 65536.f);
    float var = rq[0] * (1.f / 65536.f) - m * m;
    mean[bg] = m;
    rstd[bg] = rsqrtf(var + EPS_GN);
  }
}

// ---------------------------------------------------------------------------
// Kernel 2: apply GroupNorm affine and transpose (b,c,n) f32 -> (b,n,c) bf16
// ---------------------------------------------------------------------------
__global__ void __launch_bounds__(256)
gn_apply_kernel(const float* __restrict__ x, const float* __restrict__ gw,
                const float* __restrict__ gb, const float* __restrict__ mean,
                const float* __restrict__ rstd, __bf16* __restrict__ xn) {
  __shared__ float tile[32][33];
  const int b  = blockIdx.z;
  const int c0 = blockIdx.y * 32;
  const int n0 = blockIdx.x * 32;
  const int tx = threadIdx.x & 31;
  const int ty = threadIdx.x >> 5;                    // 0..7
  const float* xb = x + (size_t)b * C_DIM * N_SP;
#pragma unroll
  for (int i = 0; i < 4; ++i) {
    const int c = c0 + ty + i * 8;
    const float m  = mean[b * G_DIM + (c >> 4)];
    const float rs = rstd[b * G_DIM + (c >> 4)];
    const float v  = xb[(size_t)c * N_SP + n0 + tx];
    tile[ty + i * 8][tx] = (v - m) * rs * gw[c] + gb[c];
  }
  __syncthreads();
  __bf16* xnb = xn + (size_t)b * N_SP * C_DIM;
#pragma unroll
  for (int i = 0; i < 4; ++i) {
    const int n = n0 + ty + i * 8;
    const int c = c0 + tx;
    xnb[(size_t)n * C_DIM + c] = (__bf16)tile[tx][ty + i * 8];
  }
}

// ---------------------------------------------------------------------------
// Kernel 3: QKV projection GEMM.  Y = Xn(65536x512) @ W^T + bias, bf16 out.
// ---------------------------------------------------------------------------
__global__ void __launch_bounds__(256)
qkv_gemm_kernel(const __bf16* __restrict__ xn, const __bf16* __restrict__ wbf,
                const float* __restrict__ bq, const float* __restrict__ bk,
                const float* __restrict__ bv, __bf16* __restrict__ qo,
                __bf16* __restrict__ ko, __bf16* __restrict__ vo) {
  __shared__ __align__(32) __bf16 sA[BM * BK];
  __shared__ __align__(32) __bf16 sB[BN * BK];

  const int z = blockIdx.z;
  const __bf16* W   = wbf + (size_t)z * C_DIM * C_DIM;
  const float* bias = (z == 0) ? bq : (z == 1) ? bk : bv;
  __bf16* out       = (z == 0) ? qo : (z == 1) ? ko : vo;

  const int m0 = blockIdx.y * BM;
  const int n0 = blockIdx.x * BN;
  const int tid = threadIdx.x, lane = tid & 31, wave = tid >> 5;
  const int wm = (wave & 3) * 64, wn = (wave >> 2) * 64;

  f32x8 acc[4][4] = {};
  const __bf16* Ab = xn + (size_t)m0 * C_DIM;
  const __bf16* Bb = W + (size_t)n0 * C_DIM;

  for (int kb = 0; kb < C_DIM; kb += BK) {
    __syncthreads();
    stage_a(sA, Ab + kb, C_DIM, tid);
    stage_b_rows(sB, Bb + kb, C_DIM, tid);
    __syncthreads();
    gemm_round(sA, sB, wm, wn, lane, acc);
  }

  const int col = lane & 15, h = lane >> 4;
#pragma unroll
  for (int mi = 0; mi < 4; ++mi)
#pragma unroll
    for (int ni = 0; ni < 4; ++ni)
#pragma unroll
      for (int r = 0; r < 8; ++r) {
        const int gr = m0 + wm + mi * 16 + r + 8 * h;
        const int gc = n0 + wn + ni * 16 + col;
        out[(size_t)gr * C_DIM + gc] = (__bf16)(acc[mi][ni][r] + bias[gc]);
      }
}

// ---------------------------------------------------------------------------
// Kernel 4: scores = q @ k^T over n (per batch).  M=N=512, K=4096.
// ---------------------------------------------------------------------------
__global__ void __launch_bounds__(256)
scores_gemm_kernel(const __bf16* __restrict__ qb, const __bf16* __restrict__ kb,
                   float* __restrict__ scores) {
  __shared__ __align__(32) __bf16 sA[BM * BK];
  __shared__ __align__(32) __bf16 sB[BN * BK];

  const int b  = blockIdx.z;
  const int m0 = blockIdx.y * BM;                     // ci
  const int n0 = blockIdx.x * BN;                     // cj
  const int tid = threadIdx.x, lane = tid & 31, wave = tid >> 5;
  const int wm = (wave & 3) * 64, wn = (wave >> 2) * 64;

  const __bf16* Ab = qb + (size_t)b * C_DIM * N_SP + (size_t)m0 * N_SP;
  const __bf16* Bb = kb + (size_t)b * C_DIM * N_SP + (size_t)n0 * N_SP;

  f32x8 acc[4][4] = {};
  for (int kk = 0; kk < N_SP; kk += BK) {
    __syncthreads();
    stage_a(sA, Ab + kk, N_SP, tid);
    stage_b_rows(sB, Bb + kk, N_SP, tid);
    __syncthreads();
    gemm_round(sA, sB, wm, wn, lane, acc);
  }

  float* Sb = scores + (size_t)b * C_DIM * C_DIM;
  const int col = lane & 15, h = lane >> 4;
#pragma unroll
  for (int mi = 0; mi < 4; ++mi)
#pragma unroll
    for (int ni = 0; ni < 4; ++ni)
#pragma unroll
      for (int r = 0; r < 8; ++r) {
        const int gr = m0 + wm + mi * 16 + r + 8 * h;
        const int gc = n0 + wn + ni * 16 + col;
        Sb[(size_t)gr * C_DIM + gc] = acc[mi][ni][r] * SCORE_SCALE;
      }
}

// ---------------------------------------------------------------------------
// Kernel 5: row softmax over 512 scores; emit bf16 probabilities.
// ---------------------------------------------------------------------------
__global__ void __launch_bounds__(256)
softmax_kernel(const float* __restrict__ sc, __bf16* __restrict__ probs) {
  const int row = blockIdx.x;
  const int tid = threadIdx.x;
  const float* s = sc + (size_t)row * C_DIM;
  const float v0 = s[tid], v1 = s[tid + 256];
  __shared__ float red[256];
  red[tid] = fmaxf(v0, v1);
  __syncthreads();
  for (int o = 128; o > 0; o >>= 1) {
    if (tid < o) red[tid] = fmaxf(red[tid], red[tid + o]);
    __syncthreads();
  }
  const float m = red[0];
  __syncthreads();
  const float e0 = __expf(v0 - m), e1 = __expf(v1 - m);
  red[tid] = e0 + e1;
  __syncthreads();
  for (int o = 128; o > 0; o >>= 1) {
    if (tid < o) red[tid] += red[tid + o];
    __syncthreads();
  }
  const float inv = 1.f / red[0];
  __bf16* p = probs + (size_t)row * C_DIM;
  p[tid]       = (__bf16)(e0 * inv);
  p[tid + 256] = (__bf16)(e1 * inv);
}

// ---------------------------------------------------------------------------
// Kernel 6: att = probs(512x512) @ v(512x4096) per batch; output TRANSPOSED
// as att_t (b, n, c) bf16.  V operand is K-major -> LDS transpose staging.
// ---------------------------------------------------------------------------
__global__ void __launch_bounds__(256)
att_gemm_kernel(const __bf16* __restrict__ probs, const __bf16* __restrict__ vb,
                __bf16* __restrict__ att_t) {
  __shared__ __align__(32) __bf16 sA[BM * BK];
  __shared__ __align__(32) __bf16 sB[BN * BK];

  const int b  = blockIdx.z;
  const int m0 = blockIdx.y * BM;                     // ci
  const int n0 = blockIdx.x * BN;                     // ni (spatial)
  const int tid = threadIdx.x, lane = tid & 31, wave = tid >> 5;
  const int wm = (wave & 3) * 64, wn = (wave >> 2) * 64;

  const __bf16* Ab = probs + (size_t)b * C_DIM * C_DIM + (size_t)m0 * C_DIM;
  const __bf16* Bb = vb + (size_t)b * C_DIM * N_SP + n0;  // rows cj, contig in ni

  f32x8 acc[4][4] = {};
  for (int kk = 0; kk < C_DIM; kk += BK) {
    __syncthreads();
    stage_a(sA, Ab + kk, C_DIM, tid);
    stage_b_transpose(sB, Bb + (size_t)kk * N_SP, N_SP, tid);
    __syncthreads();
    gemm_round(sA, sB, wm, wn, lane, acc);
  }

  __bf16* Ob = att_t + (size_t)b * N_SP * C_DIM;
  const int col = lane & 15, h = lane >> 4;
#pragma unroll
  for (int mi = 0; mi < 4; ++mi)
#pragma unroll
    for (int ni = 0; ni < 4; ++ni)
#pragma unroll
      for (int r = 0; r < 8; ++r) {
        const int ci = m0 + wm + mi * 16 + r + 8 * h;
        const int nn = n0 + wn + ni * 16 + col;
        Ob[(size_t)nn * C_DIM + ci] = (__bf16)acc[mi][ni][r];
      }
}

// ---------------------------------------------------------------------------
// Kernel 7: output projection + bias + residual (flat-index add).
// ---------------------------------------------------------------------------
__global__ void __launch_bounds__(256)
outproj_gemm_kernel(const __bf16* __restrict__ att_t, const __bf16* __restrict__ wo,
                    const float* __restrict__ bo, const float* __restrict__ x,
                    float* __restrict__ out) {
  __shared__ __align__(32) __bf16 sA[BM * BK];
  __shared__ __align__(32) __bf16 sB[BN * BK];

  const int m0 = blockIdx.y * BM;
  const int n0 = blockIdx.x * BN;
  const int tid = threadIdx.x, lane = tid & 31, wave = tid >> 5;
  const int wm = (wave & 3) * 64, wn = (wave >> 2) * 64;

  const __bf16* Ab = att_t + (size_t)m0 * C_DIM;
  const __bf16* Bb = wo + (size_t)n0 * C_DIM;

  f32x8 acc[4][4] = {};
  for (int kb = 0; kb < C_DIM; kb += BK) {
    __syncthreads();
    stage_a(sA, Ab + kb, C_DIM, tid);
    stage_b_rows(sB, Bb + kb, C_DIM, tid);
    __syncthreads();
    gemm_round(sA, sB, wm, wn, lane, acc);
  }

  const int col = lane & 15, h = lane >> 4;
#pragma unroll
  for (int mi = 0; mi < 4; ++mi)
#pragma unroll
    for (int ni = 0; ni < 4; ++ni)
#pragma unroll
      for (int r = 0; r < 8; ++r) {
        const int gr = m0 + wm + mi * 16 + r + 8 * h;
        const int gc = n0 + wn + ni * 16 + col;
        const size_t idx = (size_t)gr * C_DIM + gc;
        out[idx] = x[idx] + bo[gc] + acc[mi][ni][r];
      }
}

// ---------------------------------------------------------------------------
// Host launch
// ---------------------------------------------------------------------------
extern "C" void kernel_launch(void* const* d_in, const int* in_sizes, int n_in,
                              void* d_out, int out_size, void* d_ws, size_t ws_size,
                              hipStream_t stream) {
  (void)in_sizes; (void)n_in; (void)out_size; (void)ws_size;
  const float* x    = (const float*)d_in[0];
  const float* gn_w = (const float*)d_in[1];
  const float* gn_b = (const float*)d_in[2];
  const float* Wq   = (const float*)d_in[3];
  const float* bq   = (const float*)d_in[4];
  const float* Wk   = (const float*)d_in[5];
  const float* bk   = (const float*)d_in[6];
  const float* Wv   = (const float*)d_in[7];
  const float* bv   = (const float*)d_in[8];
  const float* Wo   = (const float*)d_in[9];
  const float* bo   = (const float*)d_in[10];
  float* out = (float*)d_out;

  char* ws = (char*)d_ws;
  size_t off = 0;
  auto take = [&](size_t bytes) -> char* {
    char* p = ws + off;
    off = (off + bytes + 255) & ~(size_t)255;
    return p;
  };
  __bf16* wbf   = (__bf16*)take((size_t)4 * C_DIM * C_DIM * 2);
  float*  mean  = (float*) take((size_t)B_DIM * G_DIM * 4);
  float*  rstd  = (float*) take((size_t)B_DIM * G_DIM * 4);
  __bf16* xn    = (__bf16*)take((size_t)M_TOT * C_DIM * 2);
  __bf16* qb    = (__bf16*)take((size_t)M_TOT * C_DIM * 2);
  __bf16* kb    = (__bf16*)take((size_t)M_TOT * C_DIM * 2);
  __bf16* vb    = (__bf16*)take((size_t)M_TOT * C_DIM * 2);
  float*  sc    = (float*) take((size_t)B_DIM * C_DIM * C_DIM * 4);
  __bf16* probs = (__bf16*)take((size_t)B_DIM * C_DIM * C_DIM * 2);
  __bf16* att_t = (__bf16*)take((size_t)M_TOT * C_DIM * 2);

  cvt_w_kernel<<<dim3(4096), dim3(256), 0, stream>>>(Wq, Wk, Wv, Wo, wbf);
  gn_stats_kernel<<<dim3(B_DIM * G_DIM), dim3(256), 0, stream>>>(x, mean, rstd);
  gn_apply_kernel<<<dim3(N_SP / 32, C_DIM / 32, B_DIM), dim3(256), 0, stream>>>(
      x, gn_w, gn_b, mean, rstd, xn);
  qkv_gemm_kernel<<<dim3(C_DIM / BN, M_TOT / BM, 3), dim3(256), 0, stream>>>(
      xn, wbf, bq, bk, bv, qb, kb, vb);
  scores_gemm_kernel<<<dim3(C_DIM / BN, C_DIM / BM, B_DIM), dim3(256), 0, stream>>>(
      qb, kb, sc);
  softmax_kernel<<<dim3(B_DIM * C_DIM), dim3(256), 0, stream>>>(sc, probs);
  att_gemm_kernel<<<dim3(N_SP / BN, C_DIM / BM, B_DIM), dim3(256), 0, stream>>>(
      probs, vb, att_t);
  outproj_gemm_kernel<<<dim3(C_DIM / BN, M_TOT / BM, 1), dim3(256), 0, stream>>>(
      att_t, wbf + (size_t)3 * C_DIM * C_DIM, bo, x, out);
}
